// BondLengthMechanics_3212635537989
// MI455X (gfx1250) — matrix operand order repaired and checked by reading, not compiled, not run on hardware
//
#include <hip/hip_runtime.h>
#include <hip/hip_bf16.h>

typedef __attribute__((ext_vector_type(16))) _Float16 v16h;
typedef __attribute__((ext_vector_type(8)))  _Float16 h8;
typedef __attribute__((ext_vector_type(8)))  float    v8f;

#define N_NODES 100000
#define E_EDGES 500000
#define DN      128
#define DBND    64
#define DIN     384
#define LEAKY   0.001f
#define LN_EPS  1e-5f
#define DIST_MAX 20.0f
#define STEP    0.5f

#define MB      64            // edges per block (4 WMMA M-tiles)
#define NTILES  4

// padded LDS strides (stride mod 64 dwords == 4 -> conflict-free row access)
#define FSTR 392   // feat row stride (halfs)
#define HSTR 136   // hA row stride (halfs)
#define GSTR 132   // hbuf row stride (floats)

// d_ws layout (units: halfs)
#define W0T_OFF 0
#define W1T_OFF (DIN*DN)
#define W2T_OFF (DIN*DN + DN*DN)

union AF { v16h v; h8 h[2]; };

__device__ __forceinline__ float lrelu(float v) { return v >= 0.f ? v : LEAKY * v; }

// A fragment from LDS: 16-bit A 16x32 layout.
// lane<16: K = kbase+{0..7}, kbase+{16..23}; lane>=16: +8 on both runs.
__device__ __forceinline__ v16h load_A(const _Float16* base, int stride, int m, int kbase, int hi) {
    AF af;
    const _Float16* p = base + m * stride + kbase + hi * 8;
    af.h[0] = *(const h8*)(p);
    af.h[1] = *(const h8*)(p + 16);
    return af.v;
}

// B fragment from global Wt (row-major [n][K] = W transposed):
// lane<16 holds K = kbase+{0..7}, kbase+{16..23} of column n; lane>=16: +8 on both runs.
__device__ __forceinline__ v16h load_B(const _Float16* Wt, int K, int n, int kbase, int hi) {
    AF af;
    const _Float16* p = Wt + (long)n * K + kbase + hi * 8;
    af.h[0] = *(const h8*)(p);
    af.h[1] = *(const h8*)(p + 16);
    return af.v;
}

// ------- prep: transpose + fp32->fp16 convert the three weight matrices -------
__global__ void prep_weights(const float* __restrict__ W0,
                             const float* __restrict__ W1,
                             const float* __restrict__ W2,
                             _Float16* __restrict__ ws) {
    int idx = blockIdx.x * 256 + threadIdx.x;
    if (idx < DIN * DN) {            // W0t[n][k] = W0[k][n], n<128, k<384
        int n = idx / DIN, k = idx % DIN;
        ws[W0T_OFF + idx] = (_Float16)W0[k * DN + n];
    }
    if (idx < DN * DN) {             // W1t, W2t: [n][k], 128x128
        int n = idx / DN, k = idx % DN;
        ws[W1T_OFF + idx] = (_Float16)W1[k * DN + n];
        ws[W2T_OFF + idx] = (_Float16)W2[k * DN + n];
    }
}

// ------- main: 64 edges per block, 8 waves x 4 M-tiles, WMMA MLP + LN + scatter -------
__global__ __launch_bounds__(256) void edge_mlp(
    const float* __restrict__ x,
    const float* __restrict__ nodeEmb,
    const int*   __restrict__ bond_index,
    const int*   __restrict__ bond_types,
    const float* __restrict__ bondEmb,
    const _Float16* __restrict__ W0t,
    const _Float16* __restrict__ W1t,
    const _Float16* __restrict__ W2t,
    const float* __restrict__ b0,
    const float* __restrict__ b1,
    const float* __restrict__ b2,
    const float* __restrict__ ln1w, const float* __restrict__ ln1b,
    const float* __restrict__ ln2w, const float* __restrict__ ln2b,
    const float* __restrict__ Wo,  const float* __restrict__ bo,
    float* __restrict__ pos_out)
{
    // feat (64x384 f16, padded) aliased with hbuf (64x128 f32, padded):
    // feat is dead after the layer-0 WMMA loop, exactly when hbuf is first written.
    __shared__ __align__(16) char smem[MB * FSTR * sizeof(_Float16)];   // 50176 B
    _Float16* feat = (_Float16*)smem;
    float*    hbuf = (float*)smem;

    __shared__ _Float16 hA[MB * HSTR];        // 64 x 128 f16 activation (padded)
    __shared__ float    red_s[MB][4];
    __shared__ float    red_q[MB][4];
    __shared__ int      e_i[MB], e_j[MB];
    __shared__ float    e_u[MB][3];
    __shared__ float    e_dist[MB];
    __shared__ float    row_mean[MB], row_istd[MB];

    const int tid = threadIdx.x;
    const long e0 = (long)blockIdx.x * MB;

    // ---- step 1: edge geometry (clamped for tail block) ----
    if (tid < MB) {
        long e = e0 + tid; if (e >= E_EDGES) e = 0;
        int i = bond_index[2 * e];
        int j = bond_index[2 * e + 1];
        e_i[tid] = i; e_j[tid] = j;
        float dx = x[3 * i + 0] - x[3 * j + 0];
        float dy = x[3 * i + 1] - x[3 * j + 1];
        float dz = x[3 * i + 2] - x[3 * j + 2];
        float d  = sqrtf(dx * dx + dy * dy + dz * dz);
        e_dist[tid] = d;
        float inv = 1.0f / d;
        e_u[tid][0] = dx * inv; e_u[tid][1] = dy * inv; e_u[tid][2] = dz * inv;
    }
    __syncthreads();

    // ---- step 2: build feat = [emb_i | emb_j | bond_emb | smear] as f16 ----
    const float delta = DIST_MAX / (float)(DBND - 2);
    const float coeff = -0.5f / (delta * delta);

    for (int idx = tid; idx < MB * 320; idx += 256) {
        int e = idx / 320, c = idx % 320;
        float v;
        if (c < 128)      v = nodeEmb[(long)e_i[e] * DN + c];
        else if (c < 256) v = nodeEmb[(long)e_j[e] * DN + (c - 128)];
        else {
            long ge = e0 + e; if (ge >= E_EDGES) ge = 0;
            v = bondEmb[(long)bond_types[ge] * DBND + (c - 256)];
        }
        feat[e * FSTR + c] = (_Float16)v;
    }
    {   // smear: partial sums (values recomputed after reduction; no temp buffer)
        int e = tid >> 2, g = tid & 3;
        float d = e_dist[e];
        float ps = 0.f;
#pragma unroll
        for (int q = 0; q < 16; q++) {
            int c = g * 16 + q;
            float val;
            if (c < DBND - 1) { float t = d - (float)c * delta; val = __expf(coeff * t * t); }
            else              { val = (d >= DIST_MAX) ? 1.f : 0.f; }
            ps += val;
        }
        red_s[e][g] = ps;
    }
    __syncthreads();
    if (tid < MB) {
        float s = red_s[tid][0] + red_s[tid][1] + red_s[tid][2] + red_s[tid][3];
        row_istd[tid] = 1.0f / s;   // borrow row_istd as smear 1/sum
    }
    __syncthreads();
    {
        int e = tid >> 2, g = tid & 3;
        float d = e_dist[e];
        float inv = row_istd[e];
#pragma unroll
        for (int q = 0; q < 16; q++) {
            int c = g * 16 + q;
            float val;
            if (c < DBND - 1) { float t = d - (float)c * delta; val = __expf(coeff * t * t); }
            else              { val = (d >= DIST_MAX) ? 1.f : 0.f; }
            feat[e * FSTR + 320 + c] = (_Float16)(val * inv);
        }
    }
    __syncthreads();

    // ---- WMMA tiling: wave w owns output cols [16w,16w+16); 4 M-tiles each ----
    const int wv   = tid >> 5;
    const int lane = tid & 31;
    const int mrow = lane & 15;
    const int hi   = lane >> 4;
    const int nCol = wv * 16 + (lane & 15);

    v8f acc[NTILES];

    // ---- layer 0: h = lrelu(feat @ W0 + b0), K = 384 ----
#pragma unroll
    for (int t = 0; t < NTILES; t++) acc[t] = (v8f){};
#pragma unroll
    for (int kb = 0; kb < 12; kb++) {
        v16h b = load_B(W0t, DIN, nCol, kb * 32, hi);
#pragma unroll
        for (int t = 0; t < NTILES; t++) {
            v16h a = load_A(feat, FSTR, t * 16 + mrow, kb * 32, hi);
            acc[t] = __builtin_amdgcn_wmma_f32_16x16x32_f16(false, a, false, b, (short)0, acc[t], false, false);
        }
    }
    {
        float bias = b0[nCol];
#pragma unroll
        for (int t = 0; t < NTILES; t++)
#pragma unroll
            for (int r = 0; r < 8; r++) {
                float v = lrelu(acc[t][r] + bias);
                hA[(t * 16 + r + hi * 8) * HSTR + nCol] = (_Float16)v;
            }
    }
    __syncthreads();   // feat dead from here on; hbuf may alias it

    // ---- layer 1: h = lrelu(LN(h @ W1 + b1)), K = 128 ----
#pragma unroll
    for (int t = 0; t < NTILES; t++) acc[t] = (v8f){};
#pragma unroll
    for (int kb = 0; kb < 4; kb++) {
        v16h b = load_B(W1t, DN, nCol, kb * 32, hi);
#pragma unroll
        for (int t = 0; t < NTILES; t++) {
            v16h a = load_A(hA, HSTR, t * 16 + mrow, kb * 32, hi);
            acc[t] = __builtin_amdgcn_wmma_f32_16x16x32_f16(false, a, false, b, (short)0, acc[t], false, false);
        }
    }
    {
        float bias = b1[nCol];
#pragma unroll
        for (int t = 0; t < NTILES; t++)
#pragma unroll
            for (int r = 0; r < 8; r++)
                hbuf[(t * 16 + r + hi * 8) * GSTR + nCol] = acc[t][r] + bias;
    }
    __syncthreads();
    {   // LN1 stats
        int m = tid >> 2, g = tid & 3;
        float s = 0.f, q = 0.f;
#pragma unroll
        for (int c = g * 32; c < g * 32 + 32; c++) { float v = hbuf[m * GSTR + c]; s += v; q += v * v; }
        red_s[m][g] = s; red_q[m][g] = q;
    }
    __syncthreads();
    if (tid < MB) {
        float s = red_s[tid][0] + red_s[tid][1] + red_s[tid][2] + red_s[tid][3];
        float q = red_q[tid][0] + red_q[tid][1] + red_q[tid][2] + red_q[tid][3];
        float mean = s * (1.0f / 128.0f);
        float var  = q * (1.0f / 128.0f) - mean * mean;
        row_mean[tid] = mean;
        row_istd[tid] = rsqrtf(var + LN_EPS);
    }
    __syncthreads();
    for (int idx = tid; idx < MB * 128; idx += 256) {
        int m = idx >> 7, c = idx & 127;
        float v = (hbuf[m * GSTR + c] - row_mean[m]) * row_istd[m] * ln1w[c] + ln1b[c];
        hA[m * HSTR + c] = (_Float16)lrelu(v);
    }
    __syncthreads();

    // ---- layer 2: h = lrelu(LN(h @ W2 + b2)), K = 128 ----
#pragma unroll
    for (int t = 0; t < NTILES; t++) acc[t] = (v8f){};
#pragma unroll
    for (int kb = 0; kb < 4; kb++) {
        v16h b = load_B(W2t, DN, nCol, kb * 32, hi);
#pragma unroll
        for (int t = 0; t < NTILES; t++) {
            v16h a = load_A(hA, HSTR, t * 16 + mrow, kb * 32, hi);
            acc[t] = __builtin_amdgcn_wmma_f32_16x16x32_f16(false, a, false, b, (short)0, acc[t], false, false);
        }
    }
    {
        float bias = b2[nCol];
#pragma unroll
        for (int t = 0; t < NTILES; t++)
#pragma unroll
            for (int r = 0; r < 8; r++)
                hbuf[(t * 16 + r + hi * 8) * GSTR + nCol] = acc[t][r] + bias;
    }
    __syncthreads();
    {   // LN2 stats
        int m = tid >> 2, g = tid & 3;
        float s = 0.f, q = 0.f;
#pragma unroll
        for (int c = g * 32; c < g * 32 + 32; c++) { float v = hbuf[m * GSTR + c]; s += v; q += v * v; }
        red_s[m][g] = s; red_q[m][g] = q;
    }
    __syncthreads();
    if (tid < MB) {
        float s = red_s[tid][0] + red_s[tid][1] + red_s[tid][2] + red_s[tid][3];
        float q = red_q[tid][0] + red_q[tid][1] + red_q[tid][2] + red_q[tid][3];
        float mean = s * (1.0f / 128.0f);
        float var  = q * (1.0f / 128.0f) - mean * mean;
        row_mean[tid] = mean;
        row_istd[tid] = rsqrtf(var + LN_EPS);
    }
    __syncthreads();
    for (int idx = tid; idx < MB * 128; idx += 256) {
        int m = idx >> 7, c = idx & 127;
        float v = (hbuf[m * GSTR + c] - row_mean[m]) * row_istd[m] * ln2w[c] + ln2b[c];
        hbuf[m * GSTR + c] = lrelu(v);   // in-place, f32 for final GEMV
    }
    __syncthreads();

    // ---- final: force = h @ Wo + bo, scatter-add into pos_out ----
    if (tid < 2 * MB) {
        int m = tid >> 1;       // edge within tile
        int c = tid & 1;        // output column 0/1
        float s = bo[c];
#pragma unroll 8
        for (int k = 0; k < 128; k++) s += hbuf[m * GSTR + k] * Wo[k * 2 + c];
        float f = STEP * s;
        int node; float sgn;
        if (c == 0) { node = e_i[m]; sgn =  1.f; }
        else        { node = e_j[m]; sgn = -1.f; }
        if (e0 + m < E_EDGES) {
#pragma unroll
            for (int d = 0; d < 3; d++)
                atomicAdd(&pos_out[(long)node * 3 + d], sgn * f * e_u[m][d]);
        }
    }
}

extern "C" void kernel_launch(void* const* d_in, const int* in_sizes, int n_in,
                              void* d_out, int out_size, void* d_ws, size_t ws_size,
                              hipStream_t stream) {
    const float* x        = (const float*)d_in[0];
    const float* pos      = (const float*)d_in[1];
    const float* nodeEmb  = (const float*)d_in[2];
    const int*   bidx     = (const int*)  d_in[3];
    const int*   btyp     = (const int*)  d_in[4];
    const float* bemb     = (const float*)d_in[5];
    const float* W0       = (const float*)d_in[6];
    const float* b0       = (const float*)d_in[7];
    const float* W1       = (const float*)d_in[8];
    const float* b1       = (const float*)d_in[9];
    const float* ln1w     = (const float*)d_in[10];
    const float* ln1b     = (const float*)d_in[11];
    const float* W2       = (const float*)d_in[12];
    const float* b2       = (const float*)d_in[13];
    const float* ln2w     = (const float*)d_in[14];
    const float* ln2b     = (const float*)d_in[15];
    const float* Wo       = (const float*)d_in[16];
    const float* bo       = (const float*)d_in[17];
    float*       pos_out  = (float*)d_out;
    _Float16*    ws       = (_Float16*)d_ws;

    // out = input pos, then scatter-add edge updates
    hipMemcpyAsync(pos_out, pos, (size_t)out_size * sizeof(float),
                   hipMemcpyDeviceToDevice, stream);

    prep_weights<<<(DIN * DN + 255) / 256, 256, 0, stream>>>(W0, W1, W2, ws);

    int nblk = (E_EDGES + MB - 1) / MB;
    edge_mlp<<<nblk, 256, 0, stream>>>(
        x, nodeEmb, bidx, btyp, bemb,
        ws + W0T_OFF, ws + W1T_OFF, ws + W2T_OFF,
        b0, b1, b2, ln1w, ln1b, ln2w, ln2b, Wo, bo, pos_out);
}
